// BrainGCN_54408645706223
// MI455X (gfx1250) — matrix operand order repaired and checked by reading, not compiled, not run on hardware
//
#include <hip/hip_runtime.h>

// ---------------------------------------------------------------------------
// Types for CDNA5 WMMA
// ---------------------------------------------------------------------------
typedef _Float16 v16h __attribute__((ext_vector_type(16)));
typedef _Float16 v8h  __attribute__((ext_vector_type(8)));
typedef float    v8f  __attribute__((ext_vector_type(8)));

#define THREADS 256

// ---------------------------------------------------------------------------
// Utility kernels
// ---------------------------------------------------------------------------
__global__ void fill_f32_kernel(float* __restrict__ p, float v, long long n) {
    long long i = (long long)blockIdx.x * blockDim.x + threadIdx.x;
    if (i < n) p[i] = v;
}

__global__ void f32_to_f16_kernel(const float* __restrict__ in,
                                  _Float16* __restrict__ out, long long n) {
    long long i = (long long)blockIdx.x * blockDim.x + threadIdx.x;
    if (i < n) out[i] = (_Float16)in[i];
}

// deg[dst] += w  (deg pre-filled with 1.0 for the self loop)
__global__ void edge_deg_kernel(const int* __restrict__ dst,
                                const float* __restrict__ w,
                                float* __restrict__ deg, int E) {
    int e = blockIdx.x * blockDim.x + threadIdx.x;
    if (e < E) atomicAdd(&deg[dst[e]], w[e]);
}

__global__ void deg_to_dinv_kernel(float* __restrict__ d, int N) {
    int n = blockIdx.x * blockDim.x + threadIdx.x;
    if (n < N) {
        float v = d[n];
        d[n] = (v > 0.0f) ? rsqrtf(v) : 0.0f;
    }
}

// ---------------------------------------------------------------------------
// Pack W [K,64] f32 (row-major) into per-lane WMMA B-fragment order, f16.
// Flat layout: half index = kk*2048 + ct*512 + lane*16 + t
//   n    = ct*16 + (lane & 15)
//   kidx = kk*32 + (lane >> 4)*8 + (t & 7) + (t >> 3)*16
// ---------------------------------------------------------------------------
__global__ void pack_w_kernel(const float* __restrict__ W,
                              _Float16* __restrict__ Bp, int K) {
    int i = blockIdx.x * blockDim.x + threadIdx.x;
    int total = (K >> 5) * 2048;
    if (i >= total) return;
    int t    = i & 15;
    int lane = (i >> 4) & 31;
    int ct   = (i >> 9) & 3;
    int kk   = i >> 11;
    int n    = ct * 16 + (lane & 15);
    int kidx = kk * 32 + (lane >> 4) * 8 + (t & 7) + (t >> 3) * 16;
    Bp[i] = (_Float16)W[(size_t)kidx * 64 + n];
}

// ---------------------------------------------------------------------------
// GEMM: C[N,64] = A[N,K] (f16, row-major) x W (pre-packed B fragments)
// One wave computes one 16x64 tile via 4 accumulators; K-loop step 32.
// ---------------------------------------------------------------------------
__global__ __launch_bounds__(256)
void gemm_wmma_kernel(const _Float16* __restrict__ A,
                      const _Float16* __restrict__ Bp,
                      float* __restrict__ Cout, int N, int K) {
    const int lane = threadIdx.x & 31;
    const int wv   = threadIdx.x >> 5;
    const int tile = blockIdx.x * 8 + wv;      // 16-row tile index (wave-uniform)
    if (tile * 16 >= N) return;                // uniform exit: EXEC stays full

    const int r   = lane & 15;
    const int hi  = lane >> 4;                 // 0: K-sub 0..7/16..23, 1: 8..15/24..31
    int m = tile * 16 + r;
    if (m >= N) m = N - 1;                     // clamp loads for partial tail tile
    const int klo = hi * 8;
    const int nkk = K >> 5;

    const _Float16* __restrict__ arow = A + (size_t)m * K;
    const v16h* __restrict__ bv = (const v16h*)Bp;

    v8f acc0 = {}, acc1 = {}, acc2 = {}, acc3 = {};

    for (int kk = 0; kk < nkk; ++kk) {
        const _Float16* ap = arow + kk * 32 + klo;
        v8h alo = *(const v8h*)(ap);
        v8h ahi = *(const v8h*)(ap + 16);
        v16h av = __builtin_shufflevector(alo, ahi,
                    0,1,2,3,4,5,6,7,8,9,10,11,12,13,14,15);
        v16h b0 = bv[(kk * 4 + 0) * 32 + lane];
        v16h b1 = bv[(kk * 4 + 1) * 32 + lane];
        v16h b2 = bv[(kk * 4 + 2) * 32 + lane];
        v16h b3 = bv[(kk * 4 + 3) * 32 + lane];
        acc0 = __builtin_amdgcn_wmma_f32_16x16x32_f16(false, av, false, b0,
                                                      (short)0, acc0, false, false);
        acc1 = __builtin_amdgcn_wmma_f32_16x16x32_f16(false, av, false, b1,
                                                      (short)0, acc1, false, false);
        acc2 = __builtin_amdgcn_wmma_f32_16x16x32_f16(false, av, false, b2,
                                                      (short)0, acc2, false, false);
        acc3 = __builtin_amdgcn_wmma_f32_16x16x32_f16(false, av, false, b3,
                                                      (short)0, acc3, false, false);
    }

    // D layout: col = lane&15 (+16*ct), row-in-tile = hi*8 + vgpr index
    const int col   = lane & 15;
    const int rbase = hi * 8;
    for (int rr = 0; rr < 8; ++rr) {
        int row = tile * 16 + rbase + rr;
        if (row < N) {
            float* cp = Cout + (size_t)row * 64 + col;
            cp[0]  = acc0[rr];
            cp[16] = acc1[rr];
            cp[32] = acc2[rr];
            cp[48] = acc3[rr];
        }
    }
}

// ---------------------------------------------------------------------------
// Edge aggregation: agg[dst] += h[src] * (dinv[src]*w*dinv[dst]); 64 ch/edge
// ---------------------------------------------------------------------------
__global__ void edge_agg_kernel(const int* __restrict__ src,
                                const int* __restrict__ dst,
                                const float* __restrict__ w,
                                const float* __restrict__ dinv,
                                const float* __restrict__ h,
                                float* __restrict__ agg, long long total) {
    long long t = (long long)blockIdx.x * blockDim.x + threadIdx.x;
    if (t >= total) return;
    long long e = t >> 6;
    int c = (int)(t & 63);
    int s = src[e], d = dst[e];
    float norm = dinv[s] * w[e] * dinv[d];
    atomicAdd(&agg[(size_t)d * 64 + c], h[(size_t)s * 64 + c] * norm);
}

// Self loop: agg[n] += h[n] * dinv[n]^2   (sequential after edge_agg; no atomics)
__global__ void selfloop_kernel(const float* __restrict__ h,
                                const float* __restrict__ dinv,
                                float* __restrict__ agg, long long total) {
    long long i = (long long)blockIdx.x * blockDim.x + threadIdx.x;
    if (i >= total) return;
    int n = (int)(i >> 6);
    float di = dinv[n];
    agg[i] += h[i] * di * di;
}

// ---------------------------------------------------------------------------
// BatchNorm stats: stats[c] = sum, stats[64+c] = sum of squares (over N rows)
// ---------------------------------------------------------------------------
__global__ __launch_bounds__(256)
void bn_stats_kernel(const float* __restrict__ X, float* __restrict__ stats, int N) {
    __shared__ float sh[512];
    int c   = threadIdx.x & 63;
    int sub = threadIdx.x >> 6;   // 0..3
    float s0 = 0.0f, s1 = 0.0f;
    for (int n = blockIdx.x * 4 + sub; n < N; n += gridDim.x * 4) {
        float v = X[(size_t)n * 64 + c];
        s0 += v;
        s1 += v * v;
    }
    sh[threadIdx.x]       = s0;
    sh[256 + threadIdx.x] = s1;
    __syncthreads();
    if (sub == 0) {
        s0 = sh[c] + sh[64 + c] + sh[128 + c] + sh[192 + c];
        s1 = sh[256 + c] + sh[320 + c] + sh[384 + c] + sh[448 + c];
        atomicAdd(&stats[c], s0);
        atomicAdd(&stats[64 + c], s1);
    }
}

// Normalize + ReLU; write f32 (for pooling / next stage) and f16 (next GEMM)
__global__ void bn_norm_relu_kernel(const float* __restrict__ X,
                                    const float* __restrict__ stats,
                                    const float* __restrict__ gamma,
                                    const float* __restrict__ beta,
                                    float* __restrict__ Yf,
                                    _Float16* __restrict__ Yh,
                                    long long total, float invN) {
    long long i = (long long)blockIdx.x * blockDim.x + threadIdx.x;
    if (i >= total) return;
    int c = (int)(i & 63);
    float mu  = stats[c] * invN;
    float var = stats[64 + c] * invN - mu * mu;
    float s   = rsqrtf(var + 1e-5f);
    float y   = (X[i] - mu) * s * gamma[c] + beta[c];
    y = fmaxf(y, 0.0f);
    Yf[i] = y;
    Yh[i] = (_Float16)y;
}

// ---------------------------------------------------------------------------
// Pooling: per-graph mean (sum + count) and max (bit-pattern atomicMax, vals>=0)
// ---------------------------------------------------------------------------
__global__ void pool_accum_kernel(const float* __restrict__ h,
                                  const int* __restrict__ batch,
                                  float* __restrict__ psum,
                                  unsigned int* __restrict__ pmax,
                                  long long total) {
    long long i = (long long)blockIdx.x * blockDim.x + threadIdx.x;
    if (i >= total) return;
    int n = (int)(i >> 6);
    int c = (int)(i & 63);
    int g = batch[n];
    float v = h[i];
    atomicAdd(&psum[(size_t)g * 64 + c], v);
    atomicMax(&pmax[(size_t)g * 64 + c], __float_as_uint(v));
}

__global__ void pool_count_kernel(const int* __restrict__ batch,
                                  float* __restrict__ cnt, int N) {
    int n = blockIdx.x * blockDim.x + threadIdx.x;
    if (n < N) atomicAdd(&cnt[batch[n]], 1.0f);
}

// ---------------------------------------------------------------------------
// MLP head: z=[mean|max] (128) -> relu(z@fc1_w+fc1_b) (64) -> @fc2_w+fc2_b (C)
// One block (64 threads) per graph.
// ---------------------------------------------------------------------------
__global__ __launch_bounds__(64)
void head_kernel(const float* __restrict__ psum,
                 const unsigned int* __restrict__ pmax,
                 const float* __restrict__ pcnt,
                 const float* __restrict__ fc1w, const float* __restrict__ fc1b,
                 const float* __restrict__ fc2w, const float* __restrict__ fc2b,
                 float* __restrict__ out, int C) {
    __shared__ float z[128];
    __shared__ float o1[64];
    int g = blockIdx.x;
    int t = threadIdx.x;
    float cnt = fmaxf(pcnt[g], 1.0f);
    z[t]      = psum[(size_t)g * 64 + t] / cnt;
    z[64 + t] = __uint_as_float(pmax[(size_t)g * 64 + t]);
    __syncthreads();
    float a = fc1b[t];
    for (int k = 0; k < 128; ++k) a += z[k] * fc1w[(size_t)k * 64 + t];
    o1[t] = fmaxf(a, 0.0f);
    __syncthreads();
    if (t < C) {
        float b = fc2b[t];
        for (int k = 0; k < 64; ++k) b += o1[k] * fc2w[(size_t)k * C + t];
        out[(size_t)g * C + t] = b;
    }
}

// ---------------------------------------------------------------------------
// Host side
// ---------------------------------------------------------------------------
static inline unsigned int nblk(long long n) {
    return (unsigned int)((n + THREADS - 1) / THREADS);
}

extern "C" void kernel_launch(void* const* d_in, const int* in_sizes, int n_in,
                              void* d_out, int out_size, void* d_ws, size_t ws_size,
                              hipStream_t stream) {
    const float* x     = (const float*)d_in[0];
    const int*   ei    = (const int*)d_in[1];
    const float* ea    = (const float*)d_in[2];
    const int*   batch = (const int*)d_in[3];
    const float* W1 = (const float*)d_in[4];
    const float* W2 = (const float*)d_in[6];
    const float* W3 = (const float*)d_in[8];
    const float* g1 = (const float*)d_in[10];
    const float* be1 = (const float*)d_in[11];
    const float* g2 = (const float*)d_in[12];
    const float* be2 = (const float*)d_in[13];
    const float* g3 = (const float*)d_in[14];
    const float* be3 = (const float*)d_in[15];
    const float* fc1w = (const float*)d_in[16];
    const float* fc1b = (const float*)d_in[17];
    const float* fc2w = (const float*)d_in[18];
    const float* fc2b = (const float*)d_in[19];

    const int N   = in_sizes[3];
    const int E   = in_sizes[2];
    const int FIN = in_sizes[0] / N;     // 128
    const int C   = in_sizes[19];        // 2
    const int G   = out_size / C;        // 256

    const int* src = ei;
    const int* dst = ei + E;

    // Workspace layout (256B aligned blocks)
    char* ws = (char*)d_ws;
    size_t off = 0;
    auto carve = [&](size_t bytes) {
        char* p = ws + off;
        off += (bytes + 255) & ~(size_t)255;
        return p;
    };
    float*        dinv  = (float*)carve((size_t)N * 4);
    float*        hG    = (float*)carve((size_t)N * 64 * 4);   // GEMM out / act f32
    float*        hAgg  = (float*)carve((size_t)N * 64 * 4);   // aggregation buffer
    _Float16*     Xf16  = (_Float16*)carve((size_t)N * FIN * 2);
    _Float16*     Wpack = (_Float16*)carve((size_t)8192 * 2);
    float*        stats = (float*)carve(128 * 4);
    float*        psum  = (float*)carve((size_t)G * 64 * 4);
    unsigned int* pmax  = (unsigned int*)carve((size_t)G * 64 * 4);
    float*        pcnt  = (float*)carve((size_t)G * 4);
    (void)ws_size; (void)n_in;

    const long long NH = (long long)N * 64;
    const long long Etot = (long long)E * 64;

    // --- symmetric normalization coefficients -----------------------------
    fill_f32_kernel<<<nblk(N), THREADS, 0, stream>>>(dinv, 1.0f, N); // self loop
    edge_deg_kernel<<<nblk(E), THREADS, 0, stream>>>(dst, ea, dinv, E);
    deg_to_dinv_kernel<<<nblk(N), THREADS, 0, stream>>>(dinv, N);

    // --- input activations to f16 ----------------------------------------
    f32_to_f16_kernel<<<nblk((long long)N * FIN), THREADS, 0, stream>>>(
        x, Xf16, (long long)N * FIN);

    const float* Ws[3]  = {W1, W2, W3};
    const float* gs[3]  = {g1, g2, g3};
    const float* bes[3] = {be1, be2, be3};
    const int    Ksz[3] = {FIN, 64, 64};

    const int Ntiles = (N + 15) / 16;
    const unsigned int gemm_blocks = (unsigned int)((Ntiles + 7) / 8);

    for (int l = 0; l < 3; ++l) {
        const int K = Ksz[l];
        // pack weights into B-fragment order (f16)
        pack_w_kernel<<<nblk((K >> 5) * 2048), THREADS, 0, stream>>>(Ws[l], Wpack, K);
        // h = act @ W  via WMMA
        gemm_wmma_kernel<<<gemm_blocks, THREADS, 0, stream>>>(Xf16, Wpack, hG, N, K);
        // aggregation: agg = scatter-add of normalized messages + self loops
        fill_f32_kernel<<<nblk(NH), THREADS, 0, stream>>>(hAgg, 0.0f, NH);
        edge_agg_kernel<<<nblk(Etot), THREADS, 0, stream>>>(src, dst, ea, dinv,
                                                            hG, hAgg, Etot);
        selfloop_kernel<<<nblk(NH), THREADS, 0, stream>>>(hG, dinv, hAgg, NH);
        // BatchNorm + ReLU (bias cancels inside BN; b1..b3 are zero anyway)
        fill_f32_kernel<<<1, 128, 0, stream>>>(stats, 0.0f, 128);
        bn_stats_kernel<<<512, THREADS, 0, stream>>>(hAgg, stats, N);
        bn_norm_relu_kernel<<<nblk(NH), THREADS, 0, stream>>>(
            hAgg, stats, gs[l], bes[l], hG, Xf16, NH, 1.0f / (float)N);
    }

    // --- global mean / max pooling ----------------------------------------
    fill_f32_kernel<<<nblk((long long)G * 64), THREADS, 0, stream>>>(psum, 0.0f, (long long)G * 64);
    fill_f32_kernel<<<nblk((long long)G * 64), THREADS, 0, stream>>>((float*)pmax, 0.0f, (long long)G * 64);
    fill_f32_kernel<<<nblk(G), THREADS, 0, stream>>>(pcnt, 0.0f, G);
    pool_accum_kernel<<<nblk(NH), THREADS, 0, stream>>>(hG, batch, psum, pmax, NH);
    pool_count_kernel<<<nblk(N), THREADS, 0, stream>>>(batch, pcnt, N);

    // --- MLP head ----------------------------------------------------------
    head_kernel<<<G, 64, 0, stream>>>(psum, pmax, pcnt, fc1w, fc1b, fc2w, fc2b,
                                      (float*)d_out, C);
}
